// GraphEncoder_9311489098332
// MI455X (gfx1250) — compile-verified, matrix operand order checked by reading, not run on hardware
//
#include <hip/hip_runtime.h>
#include <hip/hip_bf16.h>

typedef __bf16 bf16_t;
typedef __attribute__((ext_vector_type(16))) __bf16 v16bf;
typedef __attribute__((ext_vector_type(8)))  __bf16 v8bf;
typedef __attribute__((ext_vector_type(8)))  float  v8f;

#define NNODES 100000
#define EDGES  1600000
#define EPSV   1e-5f

// ---------------------------------------------------------------------------
// Edge scatter: one wave (32 lanes) per edge, 128 floats = float4 per lane.
// Gather x[src] (coalesced b128 loads), atomic-add into agg[dst] (L2 RMW).
// Optionally counts in-degree (layer 1 only).
// ---------------------------------------------------------------------------
__global__ void __launch_bounds__(256)
scatter_accum_c128(const float* __restrict__ feat,
                   const int*   __restrict__ srcIdx,
                   const int*   __restrict__ dstIdx,
                   float*       __restrict__ agg,
                   float*       __restrict__ cnt,
                   int nEdges) {
    int edge = (int)((blockIdx.x * blockDim.x + threadIdx.x) >> 5);
    int lane = threadIdx.x & 31;
    if (edge >= nEdges) return;
    int s = srcIdx[edge];
    int d = dstIdx[edge];
    if (cnt != nullptr && lane == 0) atomicAdd(&cnt[d], 1.0f);
    const float4* fp = reinterpret_cast<const float4*>(feat + (long long)s * 128);
    float4 v = fp[lane];
    float* out = agg + (long long)d * 128 + lane * 4;
    atomicAdd(out + 0, v.x);
    atomicAdd(out + 1, v.y);
    atomicAdd(out + 2, v.z);
    atomicAdd(out + 3, v.w);
}

// ---------------------------------------------------------------------------
// Pre-pack fp32 row-major weights [K][NCOL] into bf16 WMMA B-fragment order:
// fragment (colTile ct, kStep ks): per lane l, 16 contiguous bf16 values.
//   col = ct*16 + (l&15); half = l>>4; k = ks*32 + ((v<8)?v:v+8) + 8*half
// One thread produces one lane's 32-byte packet.
// ---------------------------------------------------------------------------
template <int K, int NCOL>
__global__ void __launch_bounds__(256)
pack_weight_bf16(const float* __restrict__ W, v16bf* __restrict__ packed) {
    constexpr int total = (NCOL / 16) * (K / 32) * 32;
    int idx = blockIdx.x * blockDim.x + threadIdx.x;
    if (idx >= total) return;
    int lane  = idx & 31;
    int ks    = (idx >> 5) % (K / 32);
    int ct    = idx / (32 * (K / 32));
    int col   = ct * 16 + (lane & 15);
    int half_ = lane >> 4;
    v16bf b;
#pragma unroll
    for (int v = 0; v < 16; ++v) {
        int k = ks * 32 + ((v < 8) ? v : v + 8) + 8 * half_;
        b[v] = (bf16_t)W[k * NCOL + col];
    }
    packed[idx] = b;
}

// A fragment from LDS bf16 row (row-major, padded stride):
// 16 values = two contiguous 16-byte groups -> 2x ds_load_b128.
__device__ __forceinline__ v16bf load_a_lds(const __bf16* rowPtr, int k0, int half_) {
    v8bf lo = *reinterpret_cast<const v8bf*>(rowPtr + k0 + 8 * half_);
    v8bf hi = *reinterpret_cast<const v8bf*>(rowPtr + k0 + 16 + 8 * half_);
    return __builtin_shufflevector(lo, hi, 0, 1, 2, 3, 4, 5, 6, 7,
                                   8, 9, 10, 11, 12, 13, 14, 15);
}

// ---------------------------------------------------------------------------
// Fused SAGEConv + BN(eval) + ReLU:
//   out = relu( ((agg/cnt) @ Wl + x @ Wr + bias) * gamma/sqrt(1+eps) + beta )
// Block = 256 threads = 8 waves, covering ROWS rows x NCOL cols.
// Stage: A rows (agg pre-scaled by 1/deg, and x) -> bf16 in LDS, then each
// wave runs a 16x16 WMMA tile, K=128 in 4 steps x 2 matmuls.
// LDS row stride 136 bf16 = 68 DWORDs -> 16 lanes stripe all 64 banks.
// ---------------------------------------------------------------------------
template <int K, int NCOL, int ROWS>
__global__ void __launch_bounds__(256)
sage_gemm_bn_relu(const float* __restrict__ aggSum,
                  const float* __restrict__ cnt,
                  const float* __restrict__ xroot,
                  const v16bf* __restrict__ packWl,
                  const v16bf* __restrict__ packWr,
                  const float* __restrict__ bias,
                  const float* __restrict__ gamma,
                  const float* __restrict__ beta,
                  float* __restrict__ out) {
    constexpr int colTiles = NCOL / 16;           // 8 (layer1) or 4 (layer2)
    constexpr int LDSTRIDE = 136;                 // padded bf16 row stride
    static_assert(colTiles * (ROWS / 16) == 8, "8 waves per block");

    __shared__ __bf16 aggT[ROWS * LDSTRIDE];
    __shared__ __bf16 xT[ROWS * LDSTRIDE];

    long long nodeBase = (long long)blockIdx.x * ROWS;
    int tid = threadIdx.x;

    // ---- cooperative stage: ROWS*128 floats per matrix, float4 granules ----
    constexpr int totalVec4 = ROWS * 128 / 4;     // 512 or 1024
#pragma unroll
    for (int i = tid; i < totalVec4; i += 256) {
        int row = i >> 5;                         // 32 float4 per row
        int v4  = i & 31;
        float sc = 1.0f / fmaxf(cnt[nodeBase + row], 1.0f);
        float4 va = reinterpret_cast<const float4*>(aggSum + (nodeBase + row) * K)[v4];
        float4 vx = reinterpret_cast<const float4*>(xroot  + (nodeBase + row) * K)[v4];
        __bf16* pa = &aggT[row * LDSTRIDE + v4 * 4];
        pa[0] = (bf16_t)(va.x * sc); pa[1] = (bf16_t)(va.y * sc);
        pa[2] = (bf16_t)(va.z * sc); pa[3] = (bf16_t)(va.w * sc);
        __bf16* px = &xT[row * LDSTRIDE + v4 * 4];
        px[0] = (bf16_t)vx.x; px[1] = (bf16_t)vx.y;
        px[2] = (bf16_t)vx.z; px[3] = (bf16_t)vx.w;
    }
    __syncthreads();

    // ---- per-wave 16x16 WMMA tile ----
    int waveId = tid >> 5;
    int lane   = tid & 31;
    int half_  = lane >> 4;
    int ln16   = lane & 15;
    int rowBase = (waveId / colTiles) * 16;
    int ctile   = waveId % colTiles;
    int colN    = ctile * 16 + ln16;

    const __bf16* aggRow = &aggT[(rowBase + ln16) * LDSTRIDE];
    const __bf16* xRow   = &xT[(rowBase + ln16) * LDSTRIDE];
    const v16bf* pWl = packWl + (long long)ctile * (K / 32) * 32 + lane;
    const v16bf* pWr = packWr + (long long)ctile * (K / 32) * 32 + lane;

    v8f acc = {};
#pragma unroll
    for (int ks = 0; ks < K / 32; ++ks) {
        int k0 = ks * 32;
        v16bf aAgg = load_a_lds(aggRow, k0, half_);
        v16bf bL   = pWl[ks * 32];
        acc = __builtin_amdgcn_wmma_f32_16x16x32_bf16(
            false, aAgg, false, bL, (short)0, acc, false, false);
        v16bf aX = load_a_lds(xRow, k0, half_);
        v16bf bR = pWr[ks * 32];
        acc = __builtin_amdgcn_wmma_f32_16x16x32_bf16(
            false, aX, false, bR, (short)0, acc, false, false);
    }

    // ---- epilogue: C/D layout — VGPR r: lanes 0-15 -> M=r, 16-31 -> M=r+8 ----
    float g  = gamma[colN] * rsqrtf(1.0f + EPSV);
    float bb = bias[colN];
    float be = beta[colN];
#pragma unroll
    for (int r = 0; r < 8; ++r) {
        int mOut = rowBase + r + 8 * half_;
        float v = (acc[r] + bb) * g + be;
        v = fmaxf(v, 0.0f);
        out[(nodeBase + mOut) * NCOL + colN] = v;
    }
}

// ---------------------------------------------------------------------------
// kernel_launch: deterministic pipeline, graph-capture safe.
// ws layout: [cnt][agg: N*128 f32][h1: N*128 f32][pW1l][pW1r][pW2l][pW2r]
// ---------------------------------------------------------------------------
extern "C" void kernel_launch(void* const* d_in, const int* in_sizes, int n_in,
                              void* d_out, int out_size, void* d_ws, size_t ws_size,
                              hipStream_t stream) {
    const float* x   = (const float*)d_in[0];
    const int*   ei  = (const int*)  d_in[1];
    const float* W1l = (const float*)d_in[2];
    const float* b1  = (const float*)d_in[3];
    const float* W1r = (const float*)d_in[4];
    const float* g1  = (const float*)d_in[5];
    const float* be1 = (const float*)d_in[6];
    const float* W2l = (const float*)d_in[7];
    const float* b2  = (const float*)d_in[8];
    const float* W2r = (const float*)d_in[9];
    const float* g2  = (const float*)d_in[10];
    const float* be2 = (const float*)d_in[11];
    float* out = (float*)d_out;

    char* ws = (char*)d_ws;
    size_t off = 0;
    auto take = [&](size_t bytes) {
        char* p = ws + off;
        off = (off + bytes + 511) & ~(size_t)511;
        return p;
    };
    size_t matBytes = (size_t)NNODES * 128 * sizeof(float);
    float* cnt  = (float*)take(NNODES * sizeof(float));
    float* agg  = (float*)take(matBytes);
    float* h1   = (float*)take(matBytes);
    v16bf* pW1l = (v16bf*)take(128 * 128 * sizeof(bf16_t));
    v16bf* pW1r = (v16bf*)take(128 * 128 * sizeof(bf16_t));
    v16bf* pW2l = (v16bf*)take(128 * 64 * sizeof(bf16_t));
    v16bf* pW2r = (v16bf*)take(128 * 64 * sizeof(bf16_t));

    const int* srcIdx = ei;
    const int* dstIdx = ei + EDGES;

    dim3 blk(256);
    int scatterBlocks = EDGES / 8;  // one wave per edge, 8 waves per block

    // ---- pack weights into WMMA B-fragment bf16 layout (tiny) ----
    pack_weight_bf16<128, 128><<<4, blk, 0, stream>>>(W1l, pW1l);  // 1024 frags
    pack_weight_bf16<128, 128><<<4, blk, 0, stream>>>(W1r, pW1r);
    pack_weight_bf16<128, 64><<<2, blk, 0, stream>>>(W2l, pW2l);   // 512 frags
    pack_weight_bf16<128, 64><<<2, blk, 0, stream>>>(W2r, pW2r);

    // ---- Layer 1 ----
    hipMemsetAsync(cnt, 0, NNODES * sizeof(float), stream);
    hipMemsetAsync(agg, 0, matBytes, stream);
    scatter_accum_c128<<<scatterBlocks, blk, 0, stream>>>(
        x, srcIdx, dstIdx, agg, cnt, EDGES);
    sage_gemm_bn_relu<128, 128, 16><<<NNODES / 16, blk, 0, stream>>>(
        agg, cnt, x, pW1l, pW1r, b1, g1, be1, h1);

    // ---- Layer 2 ----
    hipMemsetAsync(agg, 0, matBytes, stream);
    scatter_accum_c128<<<scatterBlocks, blk, 0, stream>>>(
        h1, srcIdx, dstIdx, agg, nullptr, EDGES);
    sage_gemm_bn_relu<128, 64, 32><<<NNODES / 32, blk, 0, stream>>>(
        agg, cnt, h1, pW2l, pW2r, b2, g2, be2, out);
}
